// NIC_57698590655041
// MI455X (gfx1250) — compile-verified, matrix operand order checked by reading, not run on hardware
//
#include <hip/hip_runtime.h>
#include <hip/hip_bf16.h>
#include <math.h>

// ---- problem dims ----
#define B_    128
#define G_    360
#define GI_   174
#define KPAD  192     // GI padded to multiple of 32 for WMMA K-chunks
#define EF_   512
#define U_    512
#define AU_   32
#define VOC_  5000
#define VPAD  5056    // 79 * 64
#define T_    20
#define KG    1536    // gates K = EF + ET + U

// ---- CDNA5 WMMA types ----
typedef __attribute__((ext_vector_type(16))) __bf16 v16bf;
typedef __attribute__((ext_vector_type(8)))  float  v8f;

union Frag16 { v16bf v; uint4 q[2]; };

__device__ __forceinline__ unsigned short f2bf(float f) {
  union { float f; unsigned u; } x; x.f = f;
  unsigned r = x.u + 0x7fffu + ((x.u >> 16) & 1u);   // round-to-nearest-even
  return (unsigned short)(r >> 16);
}
__device__ __forceinline__ float bf2f(unsigned short h) {
  union { unsigned u; float f; } x; x.u = ((unsigned)h) << 16; return x.f;
}
__device__ __forceinline__ float leaky(float x) { return x >= 0.f ? x : 0.2f * x; }
__device__ __forceinline__ float sigm(float x)  { return 1.f / (1.f + __expf(-x)); }

// Load one 16x32 bf16 fragment from a row-major [rows][ld] bf16 matrix.
// Per CDNA5 ISA 16-bit A/B layout: lanes 0-15 hold K {k0..k0+7, k0+16..k0+23},
// lanes 16-31 hold K {k0+8..k0+15, k0+24..k0+31}; row = lane%16 within tile.
// Two contiguous 16B loads per lane (global_load_b128 / ds_load_b128).
__device__ __forceinline__ v16bf ld_frag(const unsigned short* M, int ld, int row0, int k0) {
  int lane = threadIdx.x & 31;
  int half = lane >> 4;
  const unsigned short* p = M + (size_t)(row0 + (lane & 15)) * ld + (k0 + half * 8);
  Frag16 f;
  f.q[0] = *(const uint4*)(p);
  f.q[1] = *(const uint4*)(p + 16);
  return f.v;
}

__device__ __forceinline__ v8f wmma_bf16(v16bf a, v16bf b, v8f c) {
  return __builtin_amdgcn_wmma_f32_16x16x32_bf16(false, a, false, b, (short)0, c, false, false);
}

// ======================= preprocessing kernels =======================

// enc_W [G][GI][EF] fp32 -> encWT [G][EF][KPAD] bf16 (N-major, K contiguous, zero pad)
__global__ void k_prep_encWT(const float* __restrict__ encW, unsigned short* __restrict__ o, long n) {
  long stride = (long)gridDim.x * blockDim.x;
  for (long i = (long)blockIdx.x * blockDim.x + threadIdx.x; i < n; i += stride) {
    int k = (int)(i % KPAD);
    long gn = i / KPAD;
    int nn = (int)(gn % EF_);
    long g = gn / EF_;
    float v = (k < GI_) ? encW[((size_t)g * GI_ + k) * EF_ + nn] : 0.f;
    o[i] = f2bf(v);
  }
}

// Wg [2048][1536] bf16 = [W_ih | W_hh] rows; bsum = b_ih + b_hh
__global__ void k_prep_gatesW(const float* __restrict__ W_ih, const float* __restrict__ W_hh,
                              const float* __restrict__ b_ih, const float* __restrict__ b_hh,
                              unsigned short* __restrict__ Wg, float* __restrict__ bsum) {
  long n = 2048l * KG;
  long stride = (long)gridDim.x * blockDim.x;
  long gid = (long)blockIdx.x * blockDim.x + threadIdx.x;
  for (long i = gid; i < n; i += stride) {
    int nn = (int)(i / KG);
    int k  = (int)(i % KG);
    float v = (k < 1024) ? W_ih[(size_t)nn * 1024 + k] : W_hh[(size_t)nn * 512 + (k - 1024)];
    Wg[i] = f2bf(v);
  }
  for (long i = gid; i < 2048; i += stride) bsum[i] = b_ih[i] + b_hh[i];
}

// dec_W1 [512][256] -> W1T [256][512] bf16
__global__ void k_prep_w1t(const float* __restrict__ W1, unsigned short* __restrict__ o) {
  long n = 256l * 512, stride = (long)gridDim.x * blockDim.x;
  for (long i = (long)blockIdx.x * blockDim.x + threadIdx.x; i < n; i += stride) {
    int nn = (int)(i / 512), k = (int)(i % 512);
    o[i] = f2bf(W1[(size_t)k * 256 + nn]);
  }
}

// dec_W2 [256][5000] -> W2T [5056][256] bf16 (zero-pad rows >= 5000)
__global__ void k_prep_w2t(const float* __restrict__ W2, unsigned short* __restrict__ o) {
  long n = (long)VPAD * 256, stride = (long)gridDim.x * blockDim.x;
  for (long i = (long)blockIdx.x * blockDim.x + threadIdx.x; i < n; i += stride) {
    int nn = (int)(i / 256), k = (int)(i % 256);
    float v = (nn < VOC_) ? W2[(size_t)k * VOC_ + nn] : 0.f;
    o[i] = f2bf(v);
  }
}

// xemb [T][B][512] bf16 = emb[text]
__global__ void k_prep_emb(const int* __restrict__ text, const float* __restrict__ emb,
                           unsigned short* __restrict__ o) {
  long n = (long)T_ * B_ * 512, stride = (long)gridDim.x * blockDim.x;
  for (long i = (long)blockIdx.x * blockDim.x + threadIdx.x; i < n; i += stride) {
    int t = (int)(i / (B_ * 512));
    int r = (int)(i % (B_ * 512));
    int b = r / 512, e = r % 512;
    int tok = text[b * T_ + t];
    o[i] = f2bf(emb[(size_t)tok * 512 + e]);
  }
}

// init a/c carries and the `a` slice of xin
__global__ void k_init(const float* __restrict__ a0, const float* __restrict__ c0,
                       float* __restrict__ a, float* __restrict__ c,
                       unsigned short* __restrict__ xin) {
  long n = (long)B_ * U_, stride = (long)gridDim.x * blockDim.x;
  for (long i = (long)blockIdx.x * blockDim.x + threadIdx.x; i < n; i += stride) {
    a[i] = a0[i]; c[i] = c0[i];
    int b = (int)(i / U_), u = (int)(i % U_);
    xin[(size_t)b * KG + 1024 + u] = f2bf(a0[i]);
  }
}

// ======================= encoder =======================
// One block per group g: feat_raw[g][B][EF] = leaky(x_g @ W_g + b_g) via bf16 WMMA.
__global__ __launch_bounds__(256) void k_encoder(const float* __restrict__ x,
    const unsigned short* __restrict__ encWT, const float* __restrict__ enc_b,
    unsigned short* __restrict__ feat /* [G][B][EF] bf16, pre-LN */) {
  __shared__ __align__(16) unsigned short xs[B_ * KPAD];   // 48 KB
  const int g = blockIdx.x;
  const int tid = threadIdx.x;
  for (int i = tid; i < B_ * KPAD; i += 256) xs[i] = 0;
  __syncthreads();
  const float* xg = x + (size_t)g * GI_;
  for (int i = tid; i < B_ * GI_; i += 256) {
    int b = i / GI_, k = i - b * GI_;
    xs[b * KPAD + k] = f2bf(xg[(size_t)b * (G_ * GI_) + k]);
  }
  __syncthreads();
  const unsigned short* Wg = encWT + (size_t)g * EF_ * KPAD;
  const int w = tid >> 5;            // wave id = M tile (8 waves, 8 M tiles)
  const int lane = tid & 31, half = lane >> 4, nin = lane & 15;
  for (int nc = 0; nc < 8; ++nc) {   // 8 chunks of 4 N-tiles = 512 cols
    v8f acc[4] = {};
    for (int kc = 0; kc < KPAD / 32; ++kc) {
      v16bf a = ld_frag(xs, KPAD, 16 * w, 32 * kc);
#pragma unroll
      for (int j = 0; j < 4; ++j) {
        v16bf b = ld_frag(Wg, KPAD, nc * 64 + 16 * j, 32 * kc);
        acc[j] = wmma_bf16(a, b, acc[j]);
      }
    }
#pragma unroll
    for (int j = 0; j < 4; ++j) {
      int col = nc * 64 + 16 * j + nin;
      float bias = enc_b[(size_t)g * EF_ + col];
#pragma unroll
      for (int r = 0; r < 8; ++r) {
        int row = 16 * w + r + 8 * half;
        feat[((size_t)g * B_ + row) * EF_ + col] = f2bf(leaky(acc[j][r] + bias));
      }
    }
  }
}

// LayerNorm feat rows in place + f1 = relu(feat @ attn_W1 + b1). Block per (g,b) row.
// 128 threads, 4 bf16 per thread via b64 loads/stores.
__global__ __launch_bounds__(128) void k_ln_f1(unsigned short* __restrict__ feat,
    const float* __restrict__ eg, const float* __restrict__ eb,
    const float* __restrict__ attn_W1, const float* __restrict__ attn_b1,
    float* __restrict__ f1 /* [B][G][AU] */) {
  __shared__ float buf[EF_];
  __shared__ float red[128];
  const int row = blockIdx.x;          // g*128 + b
  const int g = row >> 7, b = row & 127;
  unsigned short* fr = feat + (size_t)row * EF_;
  const int tid = threadIdx.x;
  const int e0 = tid * 4;
  float s = 0.f, s2 = 0.f;
  {
    uint2 q = *(const uint2*)(fr + e0);
    const unsigned short* us = (const unsigned short*)&q;
#pragma unroll
    for (int j = 0; j < 4; ++j) { float v = bf2f(us[j]); buf[e0 + j] = v; s += v; s2 += v * v; }
  }
  red[tid] = s; __syncthreads();
  for (int off = 64; off; off >>= 1) { if (tid < off) red[tid] += red[tid + off]; __syncthreads(); }
  float mean = red[0] * (1.f / EF_); __syncthreads();
  red[tid] = s2; __syncthreads();
  for (int off = 64; off; off >>= 1) { if (tid < off) red[tid] += red[tid + off]; __syncthreads(); }
  float var = red[0] * (1.f / EF_) - mean * mean;
  float rstd = rsqrtf(var + 1e-5f);
  __syncthreads();
  {
    uint2 q;
    unsigned short* us = (unsigned short*)&q;
#pragma unroll
    for (int j = 0; j < 4; ++j) {
      float v = (buf[e0 + j] - mean) * rstd * eg[e0 + j] + eb[e0 + j];
      buf[e0 + j] = v;
      us[j] = f2bf(v);
    }
    *(uint2*)(fr + e0) = q;
  }
  __syncthreads();
  if (tid < AU_) {
    float acc = attn_b1[tid];
    for (int e = 0; e < EF_; ++e) acc += buf[e] * attn_W1[e * AU_ + tid];
    f1[((size_t)b * G_ + g) * AU_ + tid] = fmaxf(acc, 0.f);
  }
}

// ======================= per-step kernels =======================

// attention: h2, scores, softmax over G, ctx; writes ctx+emb halves of xin and attn weights.
// ctx phase: b128 feat loads (8 bf16/lane), G-reduction split 4-ways, combined in LDS.
__global__ __launch_bounds__(256) void k_attn(const float* __restrict__ a,
    const float* __restrict__ attn_W2, const float* __restrict__ attn_b2,
    const float* __restrict__ attn_V, const float* __restrict__ attn_bV,
    const float* __restrict__ f1, const unsigned short* __restrict__ feat,
    const unsigned short* __restrict__ xemb_t, unsigned short* __restrict__ xin,
    float* __restrict__ attw, int t) {
  __shared__ float h2[AU_];
  __shared__ float sc[G_];
  __shared__ float red[256];
  __shared__ float cpart[4][EF_];      // 8 KB partial ctx
  const int b = blockIdx.x, tid = threadIdx.x;
  if (tid < AU_) {
    float acc = attn_b2[tid];
    const float* ab = a + (size_t)b * U_;
    for (int e = 0; e < U_; ++e) acc += ab[e] * attn_W2[e * AU_ + tid];
    h2[tid] = fmaxf(acc, 0.f);
  }
  __syncthreads();
  float lmax = -3.4e38f;
  for (int g = tid; g < G_; g += 256) {
    const float* fg = f1 + ((size_t)b * G_ + g) * AU_;
    float sacc = attn_bV[0];
    for (int u = 0; u < AU_; ++u) sacc += tanhf(fg[u] + h2[u]) * attn_V[u];
    sc[g] = sacc;
    lmax = fmaxf(lmax, sacc);
  }
  red[tid] = lmax; __syncthreads();
  for (int off = 128; off; off >>= 1) { if (tid < off) red[tid] = fmaxf(red[tid], red[tid + off]); __syncthreads(); }
  float mx = red[0]; __syncthreads();
  float lsum = 0.f;
  for (int g = tid; g < G_; g += 256) { float e = __expf(sc[g] - mx); sc[g] = e; lsum += e; }
  red[tid] = lsum; __syncthreads();
  for (int off = 128; off; off >>= 1) { if (tid < off) red[tid] += red[tid + off]; __syncthreads(); }
  float inv = 1.f / red[0]; __syncthreads();
  float* wout = attw + ((size_t)b * T_ + t) * G_;
  for (int g = tid; g < G_; g += 256) { float wv = sc[g] * inv; sc[g] = wv; wout[g] = wv; }
  __syncthreads();
  // ctx = sum_g w[g] * feat[g][b][:]  ---- vectorized: 64 vectors of 8 bf16, 4 g-segments
  {
    const int vt  = tid & 63;          // vector id: elements vt*8 .. vt*8+7
    const int seg = tid >> 6;          // 0..3, each covers 90 groups
    float acc[8] = {0.f, 0.f, 0.f, 0.f, 0.f, 0.f, 0.f, 0.f};
    for (int g = seg * 90; g < (seg + 1) * 90; ++g) {
      float wv = sc[g];
      uint4 q = *(const uint4*)(feat + ((size_t)g * B_ + b) * EF_ + vt * 8);
      const unsigned short* us = (const unsigned short*)&q;
#pragma unroll
      for (int j = 0; j < 8; ++j) acc[j] += wv * bf2f(us[j]);
    }
#pragma unroll
    for (int j = 0; j < 8; ++j) cpart[seg][vt * 8 + j] = acc[j];
  }
  __syncthreads();
  unsigned short* xr = xin + (size_t)b * KG;
  for (int e = tid; e < EF_; e += 256)
    xr[e] = f2bf(cpart[0][e] + cpart[1][e] + cpart[2][e] + cpart[3][e]);
  for (int e = tid; e < 512; e += 256) xr[EF_ + e] = xemb_t[(size_t)b * 512 + e];
}

// gates = xin[128x1536] @ Wg^T[1536x2048] + bsum, bf16 WMMA. grid.x = 32 (64 cols each).
__global__ __launch_bounds__(256) void k_gates(const unsigned short* __restrict__ xin,
    const unsigned short* __restrict__ Wg, const float* __restrict__ bsum,
    float* __restrict__ gates) {
  const int nb = blockIdx.x * 64;
  const int tid = threadIdx.x, w = tid >> 5, lane = tid & 31, half = lane >> 4, nin = lane & 15;
  v8f acc[4] = {};
  for (int kc = 0; kc < KG / 32; ++kc) {
    // pull next K-chunk of the weight rows toward the WGP (global_prefetch_b8)
    if (kc + 1 < KG / 32)
      __builtin_prefetch((const void*)(Wg + (size_t)(nb + (lane & 15)) * KG + 32 * (kc + 1) + half * 8), 0, 1);
    v16bf a = ld_frag(xin, KG, 16 * w, 32 * kc);
#pragma unroll
    for (int j = 0; j < 4; ++j)
      acc[j] = wmma_bf16(a, ld_frag(Wg, KG, nb + 16 * j, 32 * kc), acc[j]);
  }
#pragma unroll
  for (int j = 0; j < 4; ++j) {
    int col = nb + 16 * j + nin;
    float bias = bsum[col];
#pragma unroll
    for (int r = 0; r < 8; ++r) {
      int row = 16 * w + r + 8 * half;
      gates[(size_t)row * 2048 + col] = acc[j][r] + bias;
    }
  }
}

// LSTM cell + leaky + LayerNorm -> a; writes a fp32 and bf16 slice of xin. Block per batch.
__global__ __launch_bounds__(256) void k_cell(const float* __restrict__ gates,
    const float* __restrict__ ln_g, const float* __restrict__ ln_b,
    float* __restrict__ c, float* __restrict__ a, unsigned short* __restrict__ xin) {
  __shared__ float red[256];
  __shared__ float red2[256];
  const int b = blockIdx.x, tid = threadIdx.x;
  const float* gb = gates + (size_t)b * 2048;
  float lk[2]; float s = 0.f, s2 = 0.f;
#pragma unroll
  for (int j = 0; j < 2; ++j) {
    int u = tid + 256 * j;
    float gi = gb[u], gf = gb[512 + u], gg = gb[1024 + u], go = gb[1536 + u];
    float cv = sigm(gf) * c[(size_t)b * U_ + u] + sigm(gi) * tanhf(gg);
    c[(size_t)b * U_ + u] = cv;
    float h = sigm(go) * tanhf(cv);
    float l = leaky(h);
    lk[j] = l; s += l; s2 += l * l;
  }
  red[tid] = s; red2[tid] = s2; __syncthreads();
  for (int off = 128; off; off >>= 1) {
    if (tid < off) { red[tid] += red[tid + off]; red2[tid] += red2[tid + off]; }
    __syncthreads();
  }
  float mean = red[0] * (1.f / U_);
  float var  = red2[0] * (1.f / U_) - mean * mean;
  float rstd = rsqrtf(var + 1e-5f);
#pragma unroll
  for (int j = 0; j < 2; ++j) {
    int u = tid + 256 * j;
    float v = (lk[j] - mean) * rstd * ln_g[u] + ln_b[u];
    a[(size_t)b * U_ + u] = v;
    xin[(size_t)b * KG + 1024 + u] = f2bf(v);
  }
}

// hid = leaky(a @ dec_W1 + b1), bf16 WMMA, K=512, N=256. grid.x = 4.
__global__ __launch_bounds__(256) void k_dec1(const unsigned short* __restrict__ xin,
    const unsigned short* __restrict__ W1T, const float* __restrict__ b1,
    unsigned short* __restrict__ hid) {
  const int nb = blockIdx.x * 64;
  const int tid = threadIdx.x, w = tid >> 5, lane = tid & 31, half = lane >> 4, nin = lane & 15;
  const unsigned short* A = xin + 1024;   // `a` slice, row stride KG
  v8f acc[4] = {};
  for (int kc = 0; kc < U_ / 32; ++kc) {
    v16bf a = ld_frag(A, KG, 16 * w, 32 * kc);
#pragma unroll
    for (int j = 0; j < 4; ++j)
      acc[j] = wmma_bf16(a, ld_frag(W1T, U_, nb + 16 * j, 32 * kc), acc[j]);
  }
#pragma unroll
  for (int j = 0; j < 4; ++j) {
    int col = nb + 16 * j + nin;
    float bias = b1[col];
#pragma unroll
    for (int r = 0; r < 8; ++r) {
      int row = 16 * w + r + 8 * half;
      hid[(size_t)row * 256 + col] = f2bf(leaky(acc[j][r] + bias));
    }
  }
}

// logits = hid @ dec_W2 + b2, bf16 WMMA, K=256, N=5000 (padded 5056). grid.x = 79.
__global__ __launch_bounds__(256) void k_dec2(const unsigned short* __restrict__ hid,
    const unsigned short* __restrict__ W2T, const float* __restrict__ b2,
    float* __restrict__ logits) {
  const int nb = blockIdx.x * 64;
  const int tid = threadIdx.x, w = tid >> 5, lane = tid & 31, half = lane >> 4, nin = lane & 15;
  v8f acc[4] = {};
  for (int kc = 0; kc < 256 / 32; ++kc) {
    v16bf a = ld_frag(hid, 256, 16 * w, 32 * kc);
#pragma unroll
    for (int j = 0; j < 4; ++j)
      acc[j] = wmma_bf16(a, ld_frag(W2T, 256, nb + 16 * j, 32 * kc), acc[j]);
  }
#pragma unroll
  for (int j = 0; j < 4; ++j) {
    int col = nb + 16 * j + nin;
    if (col < VOC_) {
      float bias = b2[col];
#pragma unroll
      for (int r = 0; r < 8; ++r) {
        int row = 16 * w + r + 8 * half;
        logits[(size_t)row * VOC_ + col] = acc[j][r] + bias;
      }
    }
  }
}

// row softmax over VOC -> output[b][t][:]
__global__ __launch_bounds__(256) void k_softmax(const float* __restrict__ logits,
    float* __restrict__ out, int t) {
  __shared__ float red[256];
  const int b = blockIdx.x, tid = threadIdx.x;
  const float* lb = logits + (size_t)b * VOC_;
  float mx = -3.4e38f;
  for (int v = tid; v < VOC_; v += 256) mx = fmaxf(mx, lb[v]);
  red[tid] = mx; __syncthreads();
  for (int off = 128; off; off >>= 1) { if (tid < off) red[tid] = fmaxf(red[tid], red[tid + off]); __syncthreads(); }
  mx = red[0]; __syncthreads();
  float s = 0.f;
  for (int v = tid; v < VOC_; v += 256) s += __expf(lb[v] - mx);
  red[tid] = s; __syncthreads();
  for (int off = 128; off; off >>= 1) { if (tid < off) red[tid] += red[tid + off]; __syncthreads(); }
  float inv = 1.f / red[0];
  float* ob = out + ((size_t)b * T_ + t) * VOC_;
  for (int v = tid; v < VOC_; v += 256) ob[v] = __expf(lb[v] - mx) * inv;
}

// ======================= host launcher =======================
extern "C" void kernel_launch(void* const* d_in, const int* in_sizes, int n_in,
                              void* d_out, int out_size, void* d_ws, size_t ws_size,
                              hipStream_t stream) {
  (void)in_sizes; (void)n_in; (void)out_size; (void)ws_size;
  const float* features = (const float*)d_in[0];
  const int*   text     = (const int*)d_in[1];
  const float* a0       = (const float*)d_in[2];
  const float* c0       = (const float*)d_in[3];
  const float* enc_W    = (const float*)d_in[4];
  const float* enc_b    = (const float*)d_in[5];
  const float* enc_ln_g = (const float*)d_in[6];
  const float* enc_ln_b = (const float*)d_in[7];
  const float* attn_W1  = (const float*)d_in[8];
  const float* attn_b1  = (const float*)d_in[9];
  const float* attn_W2  = (const float*)d_in[10];
  const float* attn_b2  = (const float*)d_in[11];
  const float* attn_V   = (const float*)d_in[12];
  const float* attn_bV  = (const float*)d_in[13];
  const float* emb      = (const float*)d_in[14];
  const float* W_ih     = (const float*)d_in[15];
  const float* W_hh     = (const float*)d_in[16];
  const float* b_ih     = (const float*)d_in[17];
  const float* b_hh     = (const float*)d_in[18];
  const float* ln_g     = (const float*)d_in[19];
  const float* ln_b     = (const float*)d_in[20];
  const float* dec_W1   = (const float*)d_in[21];
  const float* dec_b1   = (const float*)d_in[22];
  const float* dec_W2   = (const float*)d_in[23];
  const float* dec_b2   = (const float*)d_in[24];

  float* out  = (float*)d_out;                       // [B][T][VOC]
  float* attw = out + (size_t)B_ * T_ * VOC_;        // [B][T][G]

  // ---- workspace carve (~140 MB) ----
  char* base = (char*)d_ws;
  size_t off = 0;
  auto carve = [&](size_t bytes) -> void* {
    off = (off + 255) & ~(size_t)255;
    void* p = base + off;
    off += bytes;
    return p;
  };
  unsigned short* encWT = (unsigned short*)carve((size_t)G_ * EF_ * KPAD * 2);   // 70.8 MB
  unsigned short* Wg    = (unsigned short*)carve((size_t)2048 * KG * 2);         // 6.3 MB
  float*          bsum  = (float*)carve(2048 * 4);
  unsigned short* W1T   = (unsigned short*)carve((size_t)256 * 512 * 2);
  unsigned short* W2T   = (unsigned short*)carve((size_t)VPAD * 256 * 2);        // 2.6 MB
  unsigned short* xemb  = (unsigned short*)carve((size_t)T_ * B_ * 512 * 2);     // 2.6 MB
  unsigned short* feat  = (unsigned short*)carve((size_t)G_ * B_ * EF_ * 2);     // 47.2 MB
  float*          f1    = (float*)carve((size_t)B_ * G_ * AU_ * 4);              // 5.9 MB
  unsigned short* xin   = (unsigned short*)carve((size_t)B_ * KG * 2);
  float*          aS    = (float*)carve((size_t)B_ * U_ * 4);
  float*          cS    = (float*)carve((size_t)B_ * U_ * 4);
  float*          gates = (float*)carve((size_t)B_ * 2048 * 4);
  unsigned short* hid   = (unsigned short*)carve((size_t)B_ * 256 * 2);
  float*          logit = (float*)carve((size_t)B_ * VOC_ * 4);

  dim3 blk(256);
  auto gs = [](long n) { return dim3((unsigned)((n + 255) / 256)); };

  // one-time (per launch) weight conversion / transposition into bf16
  k_prep_encWT <<<gs((long)G_ * EF_ * KPAD), blk, 0, stream>>>(enc_W, encWT, (long)G_ * EF_ * KPAD);
  k_prep_gatesW<<<gs(2048l * KG),            blk, 0, stream>>>(W_ih, W_hh, b_ih, b_hh, Wg, bsum);
  k_prep_w1t   <<<gs(256l * 512),            blk, 0, stream>>>(dec_W1, W1T);
  k_prep_w2t   <<<gs((long)VPAD * 256),      blk, 0, stream>>>(dec_W2, W2T);
  k_prep_emb   <<<gs((long)T_ * B_ * 512),   blk, 0, stream>>>(text, emb, xemb);
  k_init       <<<gs((long)B_ * U_),         blk, 0, stream>>>(a0, c0, aS, cS, xin);

  // encoder + LN + hidden-independent attention half
  k_encoder<<<dim3(G_),       blk,      0, stream>>>(features, encWT, enc_b, feat);
  k_ln_f1  <<<dim3(G_ * B_),  dim3(128),0, stream>>>(feat, enc_ln_g, enc_ln_b, attn_W1, attn_b1, f1);

  // recurrence: T sequential steps
  for (int t = 0; t < T_; ++t) {
    k_attn   <<<dim3(B_), blk, 0, stream>>>(aS, attn_W2, attn_b2, attn_V, attn_bV,
                                            f1, feat, xemb + (size_t)t * B_ * 512, xin, attw, t);
    k_gates  <<<dim3(32), blk, 0, stream>>>(xin, Wg, bsum, gates);
    k_cell   <<<dim3(B_), blk, 0, stream>>>(gates, ln_g, ln_b, cS, aS, xin);
    k_dec1   <<<dim3(4),  blk, 0, stream>>>(xin, W1T, dec_b1, hid);
    k_dec2   <<<dim3(79), blk, 0, stream>>>(hid, W2T, dec_b2, logit);
    k_softmax<<<dim3(B_), blk, 0, stream>>>(logit, out, t);
  }
}
